// ColaViTPreAttn_27358941675841
// MI455X (gfx1250) — matrix operand order repaired and checked by reading, not compiled, not run on hardware
//
#include <hip/hip_runtime.h>
#include <cmath>

// ColaViT pre-attention fused QKV down-projection + exact GELU.
// X:[12608,768]f32  Wq/Wk/Wv:[768,192]f32 -> 3x [12608,192]f32 (concat in d_out).
// bf16 WMMA (v_wmma_f32_16x16x32_bf16), f32 accumulate, on-the-fly f32->bf16 into LDS.
// Block tile 64x192: one column-block per output tensor (t = blockIdx.x).
// 8 waves; each wave computes a 32x48 strip = 6 WMMA tiles per K-step.

namespace {

constexpr int kM  = 12608;  // 64*197 rows
constexpr int kK  = 768;
constexpr int kR  = 192;    // rank per output tensor == BN
constexpr int BM  = 64;
constexpr int BN  = 192;
constexpr int BK  = 32;
constexpr int LDA = 40;     // padded bf16 row: 80 B = 5*16B (aligned b128, bank spread)
constexpr int LDB = 40;

typedef __attribute__((ext_vector_type(16))) __bf16   v16bf;
typedef __attribute__((ext_vector_type(8)))  float    v8f;
typedef __attribute__((ext_vector_type(4)))  unsigned u32x4;

__device__ __forceinline__ unsigned pack_bf16(float lo, float hi) {
  unsigned short a = __builtin_bit_cast(unsigned short, (__bf16)lo);
  unsigned short b = __builtin_bit_cast(unsigned short, (__bf16)hi);
  return (unsigned)a | ((unsigned)b << 16);
}

__device__ __forceinline__ float gelu_exact(float x) {
  return 0.5f * x * (1.0f + erff(x * 0.70710678118654752f));
}

} // namespace

__global__ __launch_bounds__(256)
void qkv_bf16_wmma_gelu(const float* __restrict__ X,
                        const float* __restrict__ Wq,
                        const float* __restrict__ Wk,
                        const float* __restrict__ Wv,
                        float* __restrict__ Out) {
  __shared__ __bf16 As[BM * LDA];   // A tile, row-major [m][k]
  __shared__ __bf16 Bs[BN * LDB];   // B tile, transposed [n][k]

  const int t  = blockIdx.x;        // 0..2 : which of Q/K/V (BN == rank)
  const int bm = blockIdx.y;        // 0..196
  const float* __restrict__ W = (t == 0) ? Wq : (t == 1) ? Wk : Wv;

  const int tid  = threadIdx.x;     // 0..255 (8 wave32)
  const int lane = tid & 31;
  const int wave = tid >> 5;
  const int wm   = wave & 1;        // 32-row band (2 bands)
  const int wn   = wave >> 1;       // 48-col band (4 bands)

  const int m0 = bm * BM;

  // A staging: 64x32 f32 = 2048, 8 per thread (2x float4).
  const int a_row = tid >> 2;       // 0..63
  const int a_c4  = tid & 3;        // float4 slot within first 16 cols
  // B staging: 32x192 f32 = 6144, 24 per thread (6x float4) as K-row pairs.
  const int b_p  = tid >> 4;        // 0..15 : K-row pair (rows 2p, 2p+1)
  const int b_cg = tid & 15;        // 0..15 : 12-column group

  // ISA 7.12.2 lane decomposition for 16x16 fragments.
  const int r = lane & 15;          // row (A) / col (B,C,D)
  const int h = lane >> 4;          // lane-half selects K/M sub-ranges

  v8f acc[2][3];
#pragma unroll
  for (int tm = 0; tm < 2; ++tm)
#pragma unroll
    for (int tn = 0; tn < 3; ++tn) acc[tm][tn] = v8f{};

  const float* __restrict__ asrc0 = X + (size_t)(m0 + a_row) * kK + a_c4 * 4;
  const float* __restrict__ bsrc0 = W + (size_t)(2 * b_p) * kR + b_cg * 12;

  for (int k0 = 0; k0 < kK; k0 += BK) {
    // ---- stage K-slab from global (f32) into registers ----
    const float* asrc = asrc0 + k0;
    const float* bsrc = bsrc0 + (size_t)k0 * kR;

    float4 a0 = *reinterpret_cast<const float4*>(asrc);
    float4 a1 = *reinterpret_cast<const float4*>(asrc + 16);
    float4 u0 = *reinterpret_cast<const float4*>(bsrc + 0);
    float4 u1 = *reinterpret_cast<const float4*>(bsrc + 4);
    float4 u2 = *reinterpret_cast<const float4*>(bsrc + 8);
    float4 v0 = *reinterpret_cast<const float4*>(bsrc + kR + 0);
    float4 v1 = *reinterpret_cast<const float4*>(bsrc + kR + 4);
    float4 v2 = *reinterpret_cast<const float4*>(bsrc + kR + 8);

    if (k0 + BK < kK) {
      __builtin_prefetch(asrc + BK, 0, 1);                  // global_prefetch_b8
      __builtin_prefetch(bsrc + (size_t)BK * kR, 0, 1);
    }

    __syncthreads();  // previous iteration's fragment reads complete

    // A: [m][k] bf16, packed dword stores (row stride 80B, offsets 8B-aligned)
    unsigned* ad0 = reinterpret_cast<unsigned*>(&As[a_row * LDA + a_c4 * 4]);
    ad0[0] = pack_bf16(a0.x, a0.y);
    ad0[1] = pack_bf16(a0.z, a0.w);
    unsigned* ad1 = reinterpret_cast<unsigned*>(&As[a_row * LDA + 16 + a_c4 * 4]);
    ad1[0] = pack_bf16(a1.x, a1.y);
    ad1[1] = pack_bf16(a1.z, a1.w);

    // B: transpose into [n][k]; pack (k=2p, k=2p+1) pairs -> single b32 store
    {
      const int nb = b_cg * 12;
      unsigned* bd;
      bd = reinterpret_cast<unsigned*>(&Bs[(nb + 0) * LDB + 2 * b_p]);  *bd = pack_bf16(u0.x, v0.x);
      bd = reinterpret_cast<unsigned*>(&Bs[(nb + 1) * LDB + 2 * b_p]);  *bd = pack_bf16(u0.y, v0.y);
      bd = reinterpret_cast<unsigned*>(&Bs[(nb + 2) * LDB + 2 * b_p]);  *bd = pack_bf16(u0.z, v0.z);
      bd = reinterpret_cast<unsigned*>(&Bs[(nb + 3) * LDB + 2 * b_p]);  *bd = pack_bf16(u0.w, v0.w);
      bd = reinterpret_cast<unsigned*>(&Bs[(nb + 4) * LDB + 2 * b_p]);  *bd = pack_bf16(u1.x, v1.x);
      bd = reinterpret_cast<unsigned*>(&Bs[(nb + 5) * LDB + 2 * b_p]);  *bd = pack_bf16(u1.y, v1.y);
      bd = reinterpret_cast<unsigned*>(&Bs[(nb + 6) * LDB + 2 * b_p]);  *bd = pack_bf16(u1.z, v1.z);
      bd = reinterpret_cast<unsigned*>(&Bs[(nb + 7) * LDB + 2 * b_p]);  *bd = pack_bf16(u1.w, v1.w);
      bd = reinterpret_cast<unsigned*>(&Bs[(nb + 8) * LDB + 2 * b_p]);  *bd = pack_bf16(u2.x, v2.x);
      bd = reinterpret_cast<unsigned*>(&Bs[(nb + 9) * LDB + 2 * b_p]);  *bd = pack_bf16(u2.y, v2.y);
      bd = reinterpret_cast<unsigned*>(&Bs[(nb + 10) * LDB + 2 * b_p]); *bd = pack_bf16(u2.z, v2.z);
      bd = reinterpret_cast<unsigned*>(&Bs[(nb + 11) * LDB + 2 * b_p]); *bd = pack_bf16(u2.w, v2.w);
    }

    __syncthreads();

    // ---- fragments: 2 A-tiles (reused 3x) + 3 B-tiles (reused 2x) -> 6 WMMA ----
    v16bf afrag[2];
#pragma unroll
    for (int tm = 0; tm < 2; ++tm) {
      const __bf16* ap = &As[(wm * 32 + tm * 16 + r) * LDA];
      reinterpret_cast<u32x4*>(&afrag[tm])[0] =
          *reinterpret_cast<const u32x4*>(ap + h * 8);
      reinterpret_cast<u32x4*>(&afrag[tm])[1] =
          *reinterpret_cast<const u32x4*>(ap + 16 + h * 8);
    }
    v16bf bfrag[3];
#pragma unroll
    for (int tn = 0; tn < 3; ++tn) {
      const __bf16* bp = &Bs[(wn * 48 + tn * 16 + r) * LDB + h * 8];
      reinterpret_cast<u32x4*>(&bfrag[tn])[0] = *reinterpret_cast<const u32x4*>(bp);
      reinterpret_cast<u32x4*>(&bfrag[tn])[1] = *reinterpret_cast<const u32x4*>(bp + 16);
    }

#pragma unroll
    for (int tm = 0; tm < 2; ++tm)
#pragma unroll
      for (int tn = 0; tn < 3; ++tn)
        acc[tm][tn] = __builtin_amdgcn_wmma_f32_16x16x32_bf16(
            false, afrag[tm], false, bfrag[tn], (short)0, acc[tm][tn],
            false, false);
  }

  // ---- epilogue: exact GELU, write the t-th [kM x kR] output slice ----
  // C/D layout: element i at lane L -> m_local = (L>>4)*8 + i, n_local = L&15.
  float* __restrict__ obase = Out + (size_t)t * ((size_t)kM * kR);

#pragma unroll
  for (int tm = 0; tm < 2; ++tm) {
    const int mrow0 = m0 + wm * 32 + tm * 16 + h * 8;
#pragma unroll
    for (int tn = 0; tn < 3; ++tn) {
      const int c = wn * 48 + tn * 16 + r;     // column within [0, 192)
      float* op = obase + c;
#pragma unroll
      for (int i = 0; i < 8; ++i) {
        op[(size_t)(mrow0 + i) * kR] = gelu_exact(acc[tm][tn][i]);
      }
    }
  }
}

extern "C" void kernel_launch(void* const* d_in, const int* in_sizes, int n_in,
                              void* d_out, int out_size, void* d_ws, size_t ws_size,
                              hipStream_t stream) {
  (void)in_sizes; (void)n_in; (void)out_size; (void)d_ws; (void)ws_size;
  const float* X  = (const float*)d_in[0];
  const float* Wq = (const float*)d_in[1];
  const float* Wk = (const float*)d_in[2];
  const float* Wv = (const float*)d_in[3];
  float* Out = (float*)d_out;

  dim3 grid(3 /* Q,K,V column blocks */, kM / BM /* 197 row blocks */);
  qkv_bf16_wmma_gelu<<<grid, dim3(256), 0, stream>>>(X, Wq, Wk, Wv, Out);
}
